// Decoder_67353677136089
// MI455X (gfx1250) — matrix operand order repaired and checked by reading, not compiled, not run on hardware
//
#include <hip/hip_runtime.h>
#include <hip/hip_bf16.h>

// ---------------------------------------------------------------------------
// 128-step greedy LSTM decoder for MI455X (gfx1250).
// Weights converted once per call to bf16 WMMA A-fragment layout (82.3 MB ->
// L2-resident on the 192MB L2); each serial step streams them through
// v_wmma_f32_16x16x32_bf16 matvecs (vector broadcast into all 16 B columns).
// x / h vectors are staged per-block into LDS and fed to WMMA as B-fragments
// via ds_load broadcasts. 2 kernels per step, all graph-capturable.
// ---------------------------------------------------------------------------

typedef __attribute__((ext_vector_type(16))) __bf16 v16bf;
typedef __attribute__((ext_vector_type(8)))  float  v8f;

#define HDIM   1024
#define VOCAB  32000
#define CONDN  64
#define LINN   1024
#define TSTEPS 128
#define KCH    32      // 1024 / 32 : number of K=32 chunks
#define NPART  250     // fc grid blocks (2000 tiles / 8 waves)

// workspace layout (bytes); all offsets 1KB-aligned
static const size_t WIH_OFF  = 0;                        //  8,388,608
static const size_t WHH_OFF  = 8388608;                  //  8,388,608
static const size_t FCW_OFF  = 16777216;                 // 65,536,000
static const size_t HBF_OFF  = 82313216;                 //  4,096 (2 x 1024 bf16)
static const size_t CST_OFF  = HBF_OFF + 4096;           //  4,096 (cell fp32)
static const size_t PVAL_OFF = CST_OFF + 4096;           //  1,024
static const size_t PIDX_OFF = PVAL_OFF + 1024;          //  1,024

// ---------------------------------------------------------------------------
// fp32 [rows x 1024] row-major -> bf16 WMMA 16x16x32 A-fragment layout.
// Fragment fid = tile*32 + kc : 16 rows x 32 K; lane l's 16 operand values
// stored contiguously (one 32B load at use time).
// A layout used here (contiguous K): lane<16 : M=l,      K = kc*32 + {0..15}
//                      lane>=16: M=l-16,   K = kc*32 + {16..31}
// ---------------------------------------------------------------------------
__global__ void convert_frag(const float* __restrict__ src,
                             __bf16* __restrict__ dst, int nfrag) {
  int gid  = blockIdx.x * blockDim.x + threadIdx.x;
  int fid  = gid >> 5;
  if (fid >= nfrag) return;
  int lane = gid & 31;
  int tile = fid >> 5;       // /KCH
  int kc   = fid & 31;
  int row  = tile * 16 + (lane & 15);
  int k0   = kc * 32 + ((lane & 16) ? 16 : 0);
  const float* r = src + (size_t)row * HDIM + k0;   // 32B aligned
  float4 a0 = *(const float4*)(r);
  float4 a1 = *(const float4*)(r + 4);
  float4 b0 = *(const float4*)(r + 8);
  float4 b1 = *(const float4*)(r + 12);
  v16bf f;
  f[0]  = (__bf16)a0.x; f[1]  = (__bf16)a0.y; f[2]  = (__bf16)a0.z; f[3]  = (__bf16)a0.w;
  f[4]  = (__bf16)a1.x; f[5]  = (__bf16)a1.y; f[6]  = (__bf16)a1.z; f[7]  = (__bf16)a1.w;
  f[8]  = (__bf16)b0.x; f[9]  = (__bf16)b0.y; f[10] = (__bf16)b0.z; f[11] = (__bf16)b0.w;
  f[12] = (__bf16)b1.x; f[13] = (__bf16)b1.y; f[14] = (__bf16)b1.z; f[15] = (__bf16)b1.w;
  *(v16bf*)(dst + ((size_t)fid * 512 + lane * 16)) = f;   // 32B store
}

// h0 = l1_W @ [z;c] + l1_b ; cell = 0. Wave-per-row, coalesced reads.
// Grid: 128 blocks x 256 thr (8 waves) -> 1024 waves, one row each.
__global__ void __launch_bounds__(256)
init_h0(const float* __restrict__ z, const float* __restrict__ cv,
        const float* __restrict__ l1W, const float* __restrict__ l1b,
        __bf16* __restrict__ hbf0, float* __restrict__ cstate) {
  int wv   = (blockIdx.x * blockDim.x + threadIdx.x) >> 5;  // row 0..1023
  int lane = threadIdx.x & 31;
  const float* w = l1W + (size_t)wv * (LINN + CONDN);
  float s = 0.f;
  for (int j = lane; j < LINN; j += 32)  s += w[j] * z[j];
  for (int j = lane; j < CONDN; j += 32) s += w[LINN + j] * cv[j];
#pragma unroll
  for (int off = 16; off > 0; off >>= 1) s += __shfl_xor(s, off, 32);
  if (lane == 0) {
    s += l1b[wv];
    hbf0[wv]   = (__bf16)s;
    cstate[wv] = 0.f;
  }
}

// ---------------------------------------------------------------------------
// Per-step kernel 1: finalize previous argmax -> token; stage x=relu(emb[tok])
// and h into LDS (bf16); gates = Wih@x + Whh@h + biases via WMMA; LSTM
// pointwise; write bf16 h_out. Grid: 64 blocks x 256 thr (8 waves).
// Wave w: gate quadrant w&3 (i,f,g,o), K-half w>>2. Block b owns units
// [16b, 16b+16).
// ---------------------------------------------------------------------------
__global__ void __launch_bounds__(256)
step_gates(const __bf16* __restrict__ wih, const __bf16* __restrict__ whh,
           const float* __restrict__ emb,
           const float* __restrict__ bih, const float* __restrict__ bhh,
           const float* __restrict__ pval, const int* __restrict__ pidx,
           const __bf16* __restrict__ hin, __bf16* __restrict__ hout,
           float* __restrict__ cstate, int t) {
  __shared__ __align__(32) __bf16 xs[HDIM];   // relu(emb[tok]) in bf16
  __shared__ __align__(32) __bf16 hs[HDIM];   // h in bf16
  __shared__ float gpart[8][16];
  __shared__ int   s_tok;
  int tid = threadIdx.x;

  // fused final argmax reduction (greedy feedback); t==0 -> SOS token 0
  if (tid == 0) {
    int tok = 0;
    if (t > 0) {
      float bv = pval[0]; int bi = pidx[0];
      for (int p = 1; p < NPART; ++p) {
        float v = pval[p]; int ix = pidx[p];
        if (v > bv || (v == bv && ix < bi)) { bv = v; bi = ix; }
      }
      tok = bi;
    }
    s_tok = tok;
  }
  __syncthreads();
  int tok = s_tok;

  // cooperative stage: x (4 floats/thread, relu+cvt) and h (8B/thread) -> LDS
  {
    float4 xv = *(const float4*)(emb + (size_t)tok * HDIM + tid * 4);
    xs[tid * 4 + 0] = (__bf16)(xv.x > 0.f ? xv.x : 0.f);
    xs[tid * 4 + 1] = (__bf16)(xv.y > 0.f ? xv.y : 0.f);
    xs[tid * 4 + 2] = (__bf16)(xv.z > 0.f ? xv.z : 0.f);
    xs[tid * 4 + 3] = (__bf16)(xv.w > 0.f ? xv.w : 0.f);
    ((uint2*)hs)[tid] = ((const uint2*)hin)[tid];
  }
  __syncthreads();

  int w     = tid >> 5;        // wave 0..7
  int lane  = tid & 31;
  int quad  = w & 3;           // i/f/g/o
  int khalf = w >> 2;          // K split
  int b     = blockIdx.x;      // 0..63
  int tile  = quad * 64 + b;   // tile in the 4096-row gate matrix
  int lsel  = (lane & 16) ? 16 : 0;

  v8f acc = {};
  for (int kc = khalf * 16; kc < khalf * 16 + 16; ++kc) {
    v16bf A0 = *(const v16bf*)(wih + ((size_t)(tile * KCH + kc) * 512 + lane * 16));
    v16bf A1 = *(const v16bf*)(whh + ((size_t)(tile * KCH + kc) * 512 + lane * 16));
    v16bf Bx = *(const v16bf*)(xs + kc * 32 + lsel);   // LDS broadcast
    v16bf Bh = *(const v16bf*)(hs + kc * 32 + lsel);   // LDS broadcast
    acc = __builtin_amdgcn_wmma_f32_16x16x32_bf16(false, A0, false, Bx, (short)0, acc, false, false);
    acc = __builtin_amdgcn_wmma_f32_16x16x32_bf16(false, A1, false, Bh, (short)0, acc, false, false);
  }
  // D layout: VGPR r -> lane0: M=r (N=0), lane16: M=8+r (N=0); all N equal
  if (lane == 0) {
#pragma unroll
    for (int r = 0; r < 8; ++r) gpart[w][r] = acc[r];
  } else if (lane == 16) {
#pragma unroll
    for (int r = 0; r < 8; ++r) gpart[w][8 + r] = acc[r];
  }
  __syncthreads();

  if (tid < 16) {
    int j = b * 16 + tid;
    float gi = gpart[0][tid] + gpart[4][tid] + bih[j]            + bhh[j];
    float gf = gpart[1][tid] + gpart[5][tid] + bih[HDIM + j]     + bhh[HDIM + j];
    float gg = gpart[2][tid] + gpart[6][tid] + bih[2 * HDIM + j] + bhh[2 * HDIM + j];
    float go = gpart[3][tid] + gpart[7][tid] + bih[3 * HDIM + j] + bhh[3 * HDIM + j];
    float ig = 1.f / (1.f + __expf(-gi));
    float fg = 1.f / (1.f + __expf(-gf));
    float gc = tanhf(gg);
    float og = 1.f / (1.f + __expf(-go));
    float cn = fg * cstate[j] + ig * gc;
    cstate[j] = cn;
    hout[j] = (__bf16)(og * tanhf(cn));
  }
}

// ---------------------------------------------------------------------------
// Per-step kernel 2: logits = fc_W @ h + fc_b (WMMA), write to d_out row t,
// per-block partial argmax (first-occurrence tie break).
// Grid: 250 blocks x 256 thr (8 waves); wave tile = bx*8 + w; 2000 tiles.
// ---------------------------------------------------------------------------
__global__ void __launch_bounds__(256)
step_fc(const __bf16* __restrict__ fcw, const __bf16* __restrict__ hin,
        const float* __restrict__ fcb, float* __restrict__ out,
        float* __restrict__ pval, int* __restrict__ pidx) {
  __shared__ __align__(32) __bf16 hs[HDIM];
  __shared__ float sval[128];
  __shared__ int   sidx[128];
  int tid  = threadIdx.x;
  int w    = tid >> 5;
  int lane = tid & 31;
  int tile = blockIdx.x * 8 + w;          // 0..1999
  int lsel = (lane & 16) ? 16 : 0;

  ((uint2*)hs)[tid] = ((const uint2*)hin)[tid];   // stage h once per block
  __syncthreads();

  v8f acc = {};
  for (int kc = 0; kc < KCH; ++kc) {
    v16bf A = *(const v16bf*)(fcw + ((size_t)(tile * KCH + kc) * 512 + lane * 16));
    v16bf B = *(const v16bf*)(hs + kc * 32 + lsel);   // LDS broadcast
    acc = __builtin_amdgcn_wmma_f32_16x16x32_bf16(false, A, false, B, (short)0, acc, false, false);
  }
  if (lane == 0) {
#pragma unroll
    for (int r = 0; r < 8; ++r) {
      int row = tile * 16 + r;
      float v = acc[r] + fcb[row];
      out[row] = v;
      sval[w * 16 + r] = v; sidx[w * 16 + r] = row;
    }
  } else if (lane == 16) {
#pragma unroll
    for (int r = 0; r < 8; ++r) {
      int row = tile * 16 + 8 + r;
      float v = acc[r] + fcb[row];
      out[row] = v;
      sval[w * 16 + 8 + r] = v; sidx[w * 16 + 8 + r] = row;
    }
  }
  __syncthreads();
  if (tid == 0) {
    float bv = sval[0]; int bi = sidx[0];
    for (int k = 1; k < 128; ++k) {
      if (sval[k] > bv || (sval[k] == bv && sidx[k] < bi)) { bv = sval[k]; bi = sidx[k]; }
    }
    pval[blockIdx.x] = bv;
    pidx[blockIdx.x] = bi;
  }
}

// ---------------------------------------------------------------------------
extern "C" void kernel_launch(void* const* d_in, const int* in_sizes, int n_in,
                              void* d_out, int out_size, void* d_ws, size_t ws_size,
                              hipStream_t stream) {
  const float* z   = (const float*)d_in[0];
  const float* cc  = (const float*)d_in[1];
  const float* l1W = (const float*)d_in[2];
  const float* l1b = (const float*)d_in[3];
  const float* emb = (const float*)d_in[4];
  const float* Wih = (const float*)d_in[5];
  const float* Whh = (const float*)d_in[6];
  const float* bih = (const float*)d_in[7];
  const float* bhh = (const float*)d_in[8];
  const float* fcW = (const float*)d_in[9];
  const float* fcb = (const float*)d_in[10];
  (void)in_sizes; (void)n_in; (void)out_size; (void)ws_size;

  float* out = (float*)d_out;
  char*  ws  = (char*)d_ws;
  __bf16* wihf = (__bf16*)(ws + WIH_OFF);
  __bf16* whhf = (__bf16*)(ws + WHH_OFF);
  __bf16* fcwf = (__bf16*)(ws + FCW_OFF);
  __bf16* hbf  = (__bf16*)(ws + HBF_OFF);   // double-buffered hidden (bf16)
  float*  cst  = (float*)(ws + CST_OFF);
  float*  pvl  = (float*)(ws + PVAL_OFF);
  int*    pix  = (int*)(ws + PIDX_OFF);

  // one-time (per call) weight conversion to bf16 WMMA fragments (~82MB in ws)
  convert_frag<<<1024, 256, 0, stream>>>(Wih, wihf, 8192);
  convert_frag<<<1024, 256, 0, stream>>>(Whh, whhf, 8192);
  convert_frag<<<8000, 256, 0, stream>>>(fcW, fcwf, 64000);
  init_h0<<<128, 256, 0, stream>>>(z, cc, l1W, l1b, hbf, cst);

  for (int t = 0; t < TSTEPS; ++t) {
    const __bf16* hin  = hbf + (t & 1) * HDIM;
    __bf16*       hout = hbf + ((t + 1) & 1) * HDIM;
    step_gates<<<64, 256, 0, stream>>>(wihf, whhf, emb, bih, bhh, pvl, pix,
                                       hin, hout, cst, t);
    step_fc<<<NPART, 256, 0, stream>>>(fcwf, hout, fcb,
                                       out + (size_t)t * VOCAB, pvl, pix);
  }
}
